// IterativeGNNModel_36567351558219
// MI455X (gfx1250) — matrix operand order repaired and checked by reading, not compiled, not run on hardware
//
#include <hip/hip_runtime.h>

// ---------------------------------------------------------------------------
// MI455X / gfx1250 implementation of the iterative GNN forward pass.
// Dense projections: v_wmma_f32_16x16x32_bf16 (wave32 WMMA); tiles staged with
// double-buffered global_load_async_to_lds_b128 (ASYNCcnt pipelining) from
// pre-converted bf16 activations / pre-transposed bf16 weights.
// ---------------------------------------------------------------------------

#define NN     80000      // total nodes (B * 8); multiple of 64 (tile rows)
#define BGRP   10000      // groups
#define KQDIM  512

typedef __attribute__((ext_vector_type(16))) __bf16          v16bf;
typedef __attribute__((ext_vector_type(16))) unsigned short  v16us;
typedef __attribute__((ext_vector_type(8)))  float           v8f;

__device__ __forceinline__ unsigned short f2bf(float f) {
  union { float f; unsigned u; } v; v.f = f;
  unsigned u = v.u + 0x7FFFu + ((v.u >> 16) & 1u);   // round-to-nearest-even
  return (unsigned short)(u >> 16);
}
__device__ __forceinline__ float bfdot2(unsigned a, unsigned b) {
  union { unsigned u; float f; } alo, ahi, blo, bhi;
  alo.u = a << 16; ahi.u = a & 0xFFFF0000u;
  blo.u = b << 16; bhi.u = b & 0xFFFF0000u;
  return alo.f * blo.f + ahi.f * bhi.f;
}

// f32 -> bf16 bulk convert (4 elems/thread, 16B->8B)
__global__ __launch_bounds__(256)
void k_cvt_bf16(const float* __restrict__ in, unsigned short* __restrict__ outp,
                size_t n /* multiple of 4 */)
{
  size_t i = ((size_t)blockIdx.x * blockDim.x + threadIdx.x) * 4;
  if (i >= n) return;
  float4 f = *(const float4*)(in + i);
  uint2 r;
  r.x = (unsigned)f2bf(f.x) | ((unsigned)f2bf(f.y) << 16);
  r.y = (unsigned)f2bf(f.z) | ((unsigned)f2bf(f.w) << 16);
  *(uint2*)(outp + i) = r;
}

// W[K x M] f32 -> Wt[M x K] bf16 (transpose + convert); weights are tiny
__global__ __launch_bounds__(256)
void k_cvt_wt(const float* __restrict__ W, unsigned short* __restrict__ Wt,
              int K, int M)
{
  int idx = blockIdx.x * blockDim.x + threadIdx.x;
  if (idx >= M * K) return;
  int m = idx / K, k2 = idx - m * K;
  Wt[idx] = f2bf(W[(size_t)k2 * M + m]);
}

// ---------------------------------------------------------------------------
// WMMA GEMM: C[N x M] = A[N x K] @ Wt[M x K]^T + bias; A, Wt bf16 in global,
// f32 accumulate. 128 threads = 4 waves; 64x64 tile; K step 32.
// Double-buffered async staging: each wave issues exactly 4 async b128 copies
// per tile pair (addresses clamped branchlessly; invalid chunks -> dummy LDS
// line), so "s_wait_asynccnt 4" retires exactly the previous tile while the
// next tile's copies overlap the WMMAs. LDS row stride 48 ushorts (96B,
// 16B-aligned) for the ds_load_b128 fragment reads.
// ---------------------------------------------------------------------------
#define TM 64
#define TN 64
#define TK 32
#define LSTR 48
#define BUFB (TM * LSTR * 2)   // bytes per LDS tile buffer

union FragU { uint4 q[2]; v16us v; };

__global__ __launch_bounds__(128)
void k_gemm_bf16(const unsigned short* __restrict__ A,
                 const unsigned short* __restrict__ Wt,
                 const float* __restrict__ bias, void* __restrict__ C,
                 int K, int M, int c_is_bf16)
{
  __shared__ alignas(16) unsigned short sA [2][TM][LSTR]; // [buf][row][k]
  __shared__ alignas(16) unsigned short sBt[2][TN][LSTR]; // [buf][col][k]
  __shared__ alignas(16) unsigned short sDmy[8];          // clamp target

  const int tid  = threadIdx.x;
  const int lane = tid & 31;
  const int wave = tid >> 5;                  // 0..3, rows wave*16..+15
  const int row0 = blockIdx.x * TM;
  const int col0 = blockIdx.y * TN;

  const unsigned ldsA = (unsigned)(size_t)&sA[0][0][0];
  const unsigned ldsB = (unsigned)(size_t)&sBt[0][0][0];
  const unsigned ldsD = (unsigned)(size_t)&sDmy[0];

  // issue the 4 async b128 copies for one K-step into LDS buffer `buf`.
  // Unconditional issue keeps the per-wave ASYNCcnt batch size == 4.
  auto stage = [&](int kt, int buf) {
    #pragma unroll
    for (int j = 0; j < 2; ++j) {
      int i = tid + j * 128;
      int r = i >> 2, cp = i & 3;            // A: row, 16B chunk-in-row
      bool ok = (kt + cp * 8 < K);
      unsigned d = ok ? ldsA + (unsigned)(buf * BUFB + r * (LSTR * 2) + cp * 16)
                      : ldsD;
      const unsigned short* gp =
          A + (size_t)(row0 + r) * K + (ok ? kt + cp * 8 : 0);
      asm volatile("global_load_async_to_lds_b128 %0, %1, off"
                   :: "v"(d), "v"(gp) : "memory");
    }
    #pragma unroll
    for (int j = 0; j < 2; ++j) {
      int i = tid + j * 128;
      int cc = i >> 2, cp = i & 3;           // B: col, 16B chunk-in-row
      bool ok = (col0 + cc < M) && (kt + cp * 8 < K);
      unsigned d = ok ? ldsB + (unsigned)(buf * BUFB + cc * (LSTR * 2) + cp * 16)
                      : ldsD;
      const unsigned short* gp =
          Wt + (ok ? (size_t)(col0 + cc) * K + kt + cp * 8 : (size_t)0);
      asm volatile("global_load_async_to_lds_b128 %0, %1, off"
                   :: "v"(d), "v"(gp) : "memory");
    }
  };

  v8f acc[4] = {v8f{}, v8f{}, v8f{}, v8f{}};

  const int nk = (K + TK - 1) / TK;
  stage(0, 0);                                // prologue: tile 0 in flight

  for (int it = 0; it < nk; ++it) {
    const int cur = it & 1;
    const int kt  = it * TK;

    if (it + 1 < nk) {
      stage(kt + TK, cur ^ 1);                // overlap next tile with compute
      asm volatile("s_wait_asynccnt 0x4" ::: "memory");  // retire tile `it`
    } else {
      asm volatile("s_wait_asynccnt 0x0" ::: "memory");
    }
    // zero-fill K tail (last step only; async skipped those chunks)
    if (kt + TK > K) {
      for (int i = tid; i < TM * TK; i += 128) {
        int r = i >> 5, c = i & 31;
        if (kt + c >= K) { sA[cur][r][c] = 0; sBt[cur][r][c] = 0; }
      }
    }
    __syncthreads();                          // publish tile `it` to all waves

    // A fragment (ISA 16-bit A 16x32 layout): lane m=lane&15,
    // kb = 0 (lanes 0-15) or 8 (lanes 16-31): K = {kb..kb+7, kb+16..kb+23}
    FragU ua;
    {
      const int m  = lane & 15;
      const int kb = (lane >> 4) << 3;
      ua.q[0] = *(const uint4*)&sA[cur][wave * 16 + m][kb];
      ua.q[1] = *(const uint4*)&sA[cur][wave * 16 + m][kb + 16];
    }
    v16bf afrag = __builtin_bit_cast(v16bf, ua.v);

    #pragma unroll
    for (int nt = 0; nt < 4; ++nt) {
      // B fragment (ISA 16-bit B 32x16): col n=lane&15, K = {kb2..kb2+7, kb2+16..kb2+23}
      FragU ub;
      const int n   = lane & 15;
      const int kb2 = (lane >> 4) << 3;   // 0 or 8
      ub.q[0] = *(const uint4*)&sBt[cur][nt * 16 + n][kb2];
      ub.q[1] = *(const uint4*)&sBt[cur][nt * 16 + n][kb2 + 16];
      v16bf bfrag = __builtin_bit_cast(v16bf, ub.v);

      acc[nt] = __builtin_amdgcn_wmma_f32_16x16x32_bf16(
          false, afrag, false, bfrag, (short)0, acc[nt], false, false);
    }
    __syncthreads();                          // readers done before buffer reuse
  }

  // store: C/D 16x16 f32 layout -> VGPR r: lanes0-15 M=r, lanes16-31 M=r+8
  const int n    = lane & 15;
  const int mofs = (lane >> 4) << 3;
  #pragma unroll
  for (int nt = 0; nt < 4; ++nt) {
    int gc = col0 + nt * 16 + n;
    if (gc >= M) continue;
    float bv = bias[gc];
    #pragma unroll
    for (int r = 0; r < 8; ++r) {
      int gr = row0 + wave * 16 + mofs + r;   // always < NN (64 | NN)
      float val = acc[nt][r] + bv;
      if (c_is_bf16) ((unsigned short*)C)[(size_t)gr * M + gc] = f2bf(val);
      else           ((float*)C)[(size_t)gr * M + gc] = val;
    }
  }
}

// ---------------------------------------------------------------------------
// Per-group attention. One wave per group; lane = (dst 0..3)*8 + (src 0..7).
// Only vehicle nodes (local 0..3) receive; self-edge (src==dst) masked.
// After softmax, alphas are shfl-broadcast and lanes switch to
// feature-parallel aggregation (coalesced loads/stores, no per-feature shfl).
// ---------------------------------------------------------------------------
__global__ __launch_bounds__(256)
void k_attn(const unsigned short* __restrict__ q,
            const unsigned short* __restrict__ k,
            const float* __restrict__ v,
            float* __restrict__ out, int M)
{
  const int wave = threadIdx.x >> 5;
  const int lane = threadIdx.x & 31;
  const int g    = blockIdx.x * 8 + wave;
  if (g >= BGRP) return;
  const int nb = g * 8;
  const int d  = lane >> 3;          // dst vehicle 0..3
  const int s  = lane & 7;           // src node 0..7
  const bool valid = (s != d);

  // 512-dim bf16 dot product, 16B vector loads
  const uint4* q4 = (const uint4*)(q + (size_t)(nb + d) * KQDIM);
  const uint4* k4 = (const uint4*)(k + (size_t)(nb + s) * KQDIM);
  float acc = 0.0f;
  #pragma unroll 4
  for (int c = 0; c < KQDIM / 8; ++c) {
    uint4 a = q4[c], b = k4[c];
    acc += bfdot2(a.x, b.x) + bfdot2(a.y, b.y) +
           bfdot2(a.z, b.z) + bfdot2(a.w, b.w);
  }
  float score = valid ? acc * 0.044194173824159216f /* 1/sqrt(512) */ : -1e30f;

  // softmax over the 8 lanes sharing dst d (lane xor stays within 8-group)
  float m = score;
  m = fmaxf(m, __shfl_xor(m, 1));
  m = fmaxf(m, __shfl_xor(m, 2));
  m = fmaxf(m, __shfl_xor(m, 4));
  float ex = valid ? __expf(score - m) : 0.0f;
  float den = ex;
  den += __shfl_xor(den, 1);
  den += __shfl_xor(den, 2);
  den += __shfl_xor(den, 4);
  float alpha = ex / (den + 1e-16f);

  // broadcast all pair weights to every lane (hoisted out of feature loop)
  float aAll[32];
  #pragma unroll
  for (int p = 0; p < 32; ++p) aAll[p] = __shfl(alpha, p);

  // feature-parallel aggregation: lane owns features f, f+32, ...
  for (int f = lane; f < M; f += 32) {
    float vs[8];
    #pragma unroll
    for (int s2 = 0; s2 < 8; ++s2)
      vs[s2] = v[(size_t)(nb + s2) * M + f];
    #pragma unroll
    for (int d2 = 0; d2 < 4; ++d2) {
      float sum = 0.0f;
      #pragma unroll
      for (int s2 = 0; s2 < 8; ++s2)
        sum += aAll[d2 * 8 + s2] * vs[s2];
      out[(size_t)(nb + d2) * M + f] += sum;
    }
  }
}

// ---------------------------------------------------------------------------
// BatchNorm stats: one block per column; stats[col]=sum, stats[M+col]=sumsq.
// (Each activation tensor fits in the 192MB L2, strided reads stay cheap.)
// ---------------------------------------------------------------------------
__global__ __launch_bounds__(256)
void k_bn_stats(const float* __restrict__ x, float* __restrict__ stats,
                int N, int M)
{
  const int col = blockIdx.x;
  float s = 0.0f, s2 = 0.0f;
  for (int r = threadIdx.x; r < N; r += 256) {
    float v = x[(size_t)r * M + col];
    s += v; s2 += v * v;
  }
  __shared__ float sh0[256], sh1[256];
  sh0[threadIdx.x] = s; sh1[threadIdx.x] = s2;
  __syncthreads();
  for (int o = 128; o > 0; o >>= 1) {
    if (threadIdx.x < o) {
      sh0[threadIdx.x] += sh0[threadIdx.x + o];
      sh1[threadIdx.x] += sh1[threadIdx.x + o];
    }
    __syncthreads();
  }
  if (threadIdx.x == 0) { stats[col] = sh0[0]; stats[M + col] = sh1[0]; }
}

// y = bn(x)*g+be (+res) (relu) — elementwise
__global__ __launch_bounds__(256)
void k_bn_apply(const float* __restrict__ x, const float* __restrict__ stats,
                const float* __restrict__ g, const float* __restrict__ be,
                const float* __restrict__ res, float* __restrict__ y,
                int N, int M, int do_relu)
{
  size_t i = (size_t)blockIdx.x * blockDim.x + threadIdx.x;
  if (i >= (size_t)N * M) return;
  int col = (int)(i % M);
  float inv_n = 1.0f / (float)N;
  float mean = stats[col] * inv_n;
  float var  = stats[M + col] * inv_n - mean * mean;
  float sc   = g[col] * rsqrtf(var + 1e-5f);
  float val  = (x[i] - mean) * sc + be[col];
  if (res) val += res[i];
  if (do_relu) val = fmaxf(val, 0.0f);
  y[i] = val;
}

// final: bn -> tanh -> *bound, scatter to (vehicle rows, obstacle rows)
__global__ __launch_bounds__(256)
void k_final(const float* __restrict__ x, const float* __restrict__ stats,
             const float* __restrict__ g, const float* __restrict__ be,
             float* __restrict__ out, int N)
{
  int i = blockIdx.x * blockDim.x + threadIdx.x;  // node index
  if (i >= N) return;
  int grp = i >> 3, r = i & 7;
  const float bound[2] = {1.0f, 0.8f};
  float inv_n = 1.0f / (float)N;
  #pragma unroll
  for (int c = 0; c < 2; ++c) {
    float mean = stats[c] * inv_n;
    float var  = stats[2 + c] * inv_n - mean * mean;
    float sc   = g[c] * rsqrtf(var + 1e-5f);
    float v = tanhf((x[(size_t)i * 2 + c] - mean) * sc + be[c]) * bound[c];
    size_t dst = (r < 4) ? ((size_t)(grp * 4 + r) * 2 + c)
                         : (size_t)(NN) + ((size_t)(grp * 4 + (r - 4)) * 2 + c);
    out[dst] = v;
  }
}

// ---------------------------------------------------------------------------
// Host launcher
// ---------------------------------------------------------------------------
struct ConvW { const float *Wq,*bq,*Wk,*bk,*Wv,*bv,*Ws,*bs; };

static inline ConvW convw(void* const* d, int b) {
  // jax pytree (alphabetical): Wk, Wq, Ws, Wv, bk, bq, bs, bv
  ConvW w;
  w.Wk = (const float*)d[b+0]; w.Wq = (const float*)d[b+1];
  w.Ws = (const float*)d[b+2]; w.Wv = (const float*)d[b+3];
  w.bk = (const float*)d[b+4]; w.bq = (const float*)d[b+5];
  w.bs = (const float*)d[b+6]; w.bv = (const float*)d[b+7];
  return w;
}

extern "C" void kernel_launch(void* const* d_in, const int* in_sizes, int n_in,
                              void* d_out, int out_size, void* d_ws, size_t ws_size,
                              hipStream_t stream)
{
  (void)in_sizes; (void)n_in; (void)out_size; (void)ws_size;
  // Input order: x0, then params flattened as jax pytree (sorted keys):
  //   b0{W,b,be,g}, b3{W,b,be,g},
  //   blk1{be1,be2,c1{Wk,Wq,Ws,Wv,bk,bq,bs,bv},c2{...},g1,g2}, blk2{...},
  //   edges, vehicle_idx, obstacle_idx
  const float* x0    = (const float*)d_in[0];
  const float* b0_W  = (const float*)d_in[1];
  const float* b0_b  = (const float*)d_in[2];
  const float* b0_be = (const float*)d_in[3];
  const float* b0_g  = (const float*)d_in[4];
  const float* b3_W  = (const float*)d_in[5];
  const float* b3_b  = (const float*)d_in[6];
  const float* b3_be = (const float*)d_in[7];
  const float* b3_g  = (const float*)d_in[8];

  struct Blk { const float *be1,*be2,*g1,*g2; ConvW c1,c2; };
  auto blk = [&](int base) {
    Blk b;
    b.be1 = (const float*)d_in[base + 0];
    b.be2 = (const float*)d_in[base + 1];
    b.c1  = convw(d_in, base + 2);
    b.c2  = convw(d_in, base + 10);
    b.g1  = (const float*)d_in[base + 18];
    b.g2  = (const float*)d_in[base + 19];
    return b;
  };
  Blk B1 = blk(9), B2 = blk(29);

  // workspace carve (aligned)
  char* ws = (char*)d_ws; size_t off = 0;
  auto carve = [&](size_t bytes) -> void* {
    void* p = ws + off; off += (bytes + 255) & ~(size_t)255; return p;
  };
  float*          xA    = (float*)carve((size_t)NN * 160 * 4);
  float*          xB    = (float*)carve((size_t)NN * 160 * 4);
  unsigned short* xbf   = (unsigned short*)carve((size_t)NN * 160 * 2);
  unsigned short* qb    = (unsigned short*)carve((size_t)NN * KQDIM * 2);
  unsigned short* kb    = (unsigned short*)carve((size_t)NN * KQDIM * 2);
  float*          vv    = (float*)carve((size_t)NN * 160 * 4);
  float*          so    = (float*)carve((size_t)NN * 160 * 4);
  unsigned short* wt    = (unsigned short*)carve((size_t)512 * 160 * 2);
  float*          stats = (float*)carve(4096);

  // launch helpers (all on `stream`, sequential -> wt/xbf reuse is safe)
  auto cvt_x = [&](const float* x, int K) {
    size_t n = (size_t)NN * K;
    k_cvt_bf16<<<(unsigned)((n / 4 + 255) / 256), 256, 0, stream>>>(x, xbf, n);
  };
  auto gemm = [&](const float* W, const float* bias, void* C, int K, int M,
                  int cbf16) {
    k_cvt_wt<<<(M * K + 255) / 256, 256, 0, stream>>>(W, wt, K, M);
    dim3 grid(NN / TM, (M + TN - 1) / TN);
    k_gemm_bf16<<<grid, 128, 0, stream>>>(xbf, wt, bias, C, K, M, cbf16);
  };
  auto bn_apply = [&](const float* x, const float* g, const float* be,
                      const float* res, float* y, int M, int relu) {
    size_t tot = (size_t)NN * M;
    k_bn_apply<<<(unsigned)((tot + 255) / 256), 256, 0, stream>>>(
        x, stats, g, be, res, y, NN, M, relu);
  };
  auto conv = [&](const float* x, int din, int dout, const ConvW& w) {
    cvt_x(x, din);
    gemm(w.Wq, w.bq, qb, din, KQDIM, 1);
    gemm(w.Wk, w.bk, kb, din, KQDIM, 1);
    gemm(w.Wv, w.bv, vv, din, dout, 0);
    gemm(w.Ws, w.bs, so, din, dout, 0);
    k_attn<<<BGRP / 8, 256, 0, stream>>>(qb, kb, vv, so, dout);
  };

  // layer 0: xB = x0 @ W0 + b0 -> bn -> relu -> xA  (N x 80)
  cvt_x(x0, 8);
  gemm(b0_W, b0_b, xB, 8, 80, 0);
  k_bn_stats<<<80, 256, 0, stream>>>(xB, stats, NN, 80);
  bn_apply(xB, b0_g, b0_be, nullptr, xA, 80, 1);

  // block 1
  conv(xA, 80, 160, B1.c1);
  k_bn_stats<<<160, 256, 0, stream>>>(so, stats, NN, 160);
  bn_apply(so, B1.g1, B1.be1, nullptr, xB, 160, 1);
  conv(xB, 160, 80, B1.c2);
  k_bn_stats<<<80, 256, 0, stream>>>(so, stats, NN, 80);
  bn_apply(so, B1.g2, B1.be2, xA, xA, 80, 1);   // +residual, relu, in-place safe

  // block 2
  conv(xA, 80, 160, B2.c1);
  k_bn_stats<<<160, 256, 0, stream>>>(so, stats, NN, 160);
  bn_apply(so, B2.g1, B2.be1, nullptr, xB, 160, 1);
  conv(xB, 160, 80, B2.c2);
  k_bn_stats<<<80, 256, 0, stream>>>(so, stats, NN, 80);
  bn_apply(so, B2.g2, B2.be2, xA, xA, 80, 1);

  // head: xB = xA @ W3 + b3 (N x 2) -> bn -> tanh -> bound -> scatter
  cvt_x(xA, 80);
  gemm(b3_W, b3_b, xB, 80, 2, 0);
  k_bn_stats<<<2, 256, 0, stream>>>(xB, stats, NN, 2);
  k_final<<<(NN + 255) / 256, 256, 0, stream>>>(xB, stats, b3_g, b3_be,
                                                (float*)d_out, NN);
}
